// ParallelExperts_6451040878924
// MI455X (gfx1250) — matrix operand order repaired and checked by reading, not compiled, not run on hardware
//
#include <hip/hip_runtime.h>
#include <hip/hip_bf16.h>
#include <stdint.h>

typedef __attribute__((ext_vector_type(16))) _Float16 v16h;
typedef __attribute__((ext_vector_type(8)))  _Float16 v8h;
typedef __attribute__((ext_vector_type(4)))  _Float16 v4h;
typedef __attribute__((ext_vector_type(8)))  float    v8f;
typedef __attribute__((ext_vector_type(4)))  int      v4i;

#define D_IN  1024
#define D_OUT 1024
#define N_EXP 8
#define D_HID 2048
#define N_TOK 4096

// gfx1250 async global->LDS staging (ASYNCcnt path). Guarded so the file
// still compiles on the host pass / toolchains lacking the builtins.
#if __has_builtin(__builtin_amdgcn_global_load_async_to_lds_b128) && \
    __has_builtin(__builtin_amdgcn_s_wait_asynccnt)
#define USE_ASYNC 1
#else
#define USE_ASYNC 0
#endif

#if USE_ASYNC
typedef __attribute__((address_space(1))) v4i gas_v4i;  // global
typedef __attribute__((address_space(3))) v4i las_v4i;  // LDS
__device__ __forceinline__ void async_copy16(const _Float16* g, _Float16* l) {
  __builtin_amdgcn_global_load_async_to_lds_b128((gas_v4i*)g, (las_v4i*)l, 0, 0);
}
#endif

// ---------------- elementwise f32 -> f16 (4 elems / thread) ----------------
__global__ void k_f32_to_f16(const float* __restrict__ s, _Float16* __restrict__ d, int n4) {
  int i = blockIdx.x * blockDim.x + threadIdx.x;
  if (i < n4) {
    float4 v = reinterpret_cast<const float4*>(s)[i];
    v4h h;
    h[0] = (_Float16)v.x; h[1] = (_Float16)v.y;
    h[2] = (_Float16)v.z; h[3] = (_Float16)v.w;
    reinterpret_cast<v4h*>(d)[i] = h;
  }
}

// ------- tiled transpose + convert: src f32 [R][C] -> dst f16 [C][R] -------
__global__ void k_transpose_convert(const float* __restrict__ src,
                                    _Float16* __restrict__ dst, int R, int C) {
  const size_t bofs = (size_t)blockIdx.z * R * C;
  src += bofs; dst += bofs;
  __shared__ float t[32][33];
  const int c0 = blockIdx.x * 32, r0 = blockIdx.y * 32;
  const int tx = threadIdx.x, ty = threadIdx.y;
  for (int j = ty; j < 32; j += 8)
    t[j][tx] = src[(size_t)(r0 + j) * C + (c0 + tx)];
  __syncthreads();
  for (int j = ty; j < 32; j += 8)
    dst[(size_t)(c0 + j) * R + (r0 + tx)] = (_Float16)t[tx][j];
}

// -------------- ybar init: ybar[n][o] = mean_e b2[e][o] --------------------
__global__ void k_bias_init(const float* __restrict__ b2, float* __restrict__ ybar, int total) {
  int i = blockIdx.x * blockDim.x + threadIdx.x;
  if (i < total) {
    int o = i & (D_OUT - 1);
    float s = 0.f;
#pragma unroll
    for (int e = 0; e < N_EXP; ++e) s += b2[e * D_OUT + o];
    ybar[i] = s * (1.0f / N_EXP);
  }
}

// ---------------------------- WMMA GEMM ------------------------------------
// C[M,N] (+)= A[M,K] * B^T[N,K]   (A,Bt are f16; Bt stored transposed)
// MODE 0: outH = (f16) relu(acc + bias[n])          (GEMM1, H per expert)
// MODE 1: outF += scale * acc                       (GEMM2, accumulate ybar)
// MODE 2: outF  = acc + bias[n]                     (final linear)
template <int MODE>
__global__ __launch_bounds__(256) void k_gemm(
    const _Float16* __restrict__ A, int lda,
    const _Float16* __restrict__ Bt, int ldb,
    const float* __restrict__ bias,
    _Float16* __restrict__ outH, float* __restrict__ outF, int ldo,
    int K, float scale) {
  constexpr int BM = 128, BN = 128, BK = 32;
  constexpr int LDS_LD = BK + 8;  // 40 halves = 80B stride -> conflict-free b128
  __shared__ __attribute__((aligned(16))) _Float16 As[2][BM * LDS_LD];
  __shared__ __attribute__((aligned(16))) _Float16 Bs[2][BN * LDS_LD];

  const int tid   = threadIdx.x;
  const int wave  = tid >> 5, lane = tid & 31;
  const int waveM = wave >> 2, waveN = wave & 3;   // 2 x 4 waves
  const int lrow  = lane & 15, lhi = lane >> 4;
  const int bm0   = blockIdx.y * BM, bn0 = blockIdx.x * BN;

  // staging: 512 x 16B chunks per tile, 2 per thread (4 chunks per 64B row)
  const int id0 = tid, id1 = tid + 256;
  const int row0 = id0 >> 2, seg0 = id0 & 3;
  const int row1 = id1 >> 2, seg1 = id1 & 3;

  const _Float16* gA0 = A  + (size_t)(bm0 + row0) * lda + seg0 * 8;
  const _Float16* gA1 = A  + (size_t)(bm0 + row1) * lda + seg1 * 8;
  const _Float16* gB0 = Bt + (size_t)(bn0 + row0) * ldb + seg0 * 8;
  const _Float16* gB1 = Bt + (size_t)(bn0 + row1) * ldb + seg1 * 8;
  const int sO0 = row0 * LDS_LD + seg0 * 8;
  const int sO1 = row1 * LDS_LD + seg1 * 8;

  v8f acc[4][2] = {};
  const int nk = K / BK;

#if USE_ASYNC
  // ---- prologue: DMA tile 0 straight into LDS buffer 0 ----
  async_copy16(gA0, &As[0][sO0]);
  async_copy16(gA1, &As[0][sO1]);
  async_copy16(gB0, &Bs[0][sO0]);
  async_copy16(gB1, &Bs[0][sO1]);
#else
  v8h ra0 = *(const v8h*)gA0, ra1 = *(const v8h*)gA1;
  v8h rb0 = *(const v8h*)gB0, rb1 = *(const v8h*)gB1;
  *(v8h*)&As[0][sO0] = ra0;  *(v8h*)&As[0][sO1] = ra1;
  *(v8h*)&Bs[0][sO0] = rb0;  *(v8h*)&Bs[0][sO1] = rb1;
#endif

  for (int kk = 0; kk < nk; ++kk) {
#if USE_ASYNC
    __builtin_amdgcn_s_wait_asynccnt(0);  // my DMA writes landed in LDS
#endif
    __syncthreads();                      // => everyone's writes landed
    const int cur = kk & 1;
    const bool more = (kk + 1) < nk;
    if (more) {
      const int off = (kk + 1) * BK;
      const int nxt = 1 - cur;
#if USE_ASYNC
      // DMA next tile while the matrix pipe chews on buf[cur]
      async_copy16(gA0 + off, &As[nxt][sO0]);
      async_copy16(gA1 + off, &As[nxt][sO1]);
      async_copy16(gB0 + off, &Bs[nxt][sO0]);
      async_copy16(gB1 + off, &Bs[nxt][sO1]);
#else
      ra0 = *(const v8h*)(gA0 + off); ra1 = *(const v8h*)(gA1 + off);
      rb0 = *(const v8h*)(gB0 + off); rb1 = *(const v8h*)(gB1 + off);
#endif
      if (kk + 2 < nk) {  // gfx1250 global_prefetch_b8 two tiles ahead
        __builtin_prefetch(gA0 + off + BK, 0, 1);
        __builtin_prefetch(gB0 + off + BK, 0, 1);
      }
    }

    // A fragments: lane=row M, halves select K octets (ISA 16-bit A 16x32 layout)
    v16h af[4], bfr[2];
#pragma unroll
    for (int mt = 0; mt < 4; ++mt) {
      const _Float16* p = &As[cur][(waveM * 64 + mt * 16 + lrow) * LDS_LD];
      v8h lo = *(const v8h*)(p + lhi * 8);
      v8h hi = *(const v8h*)(p + 16 + lhi * 8);
      af[mt] = __builtin_shufflevector(lo, hi, 0, 1, 2, 3, 4, 5, 6, 7,
                                       8, 9, 10, 11, 12, 13, 14, 15);
    }
    // B fragments: lane=col N, halves select K octets (same as A); Bs is [N][K]
#pragma unroll
    for (int nt = 0; nt < 2; ++nt) {
      const _Float16* p = &Bs[cur][(waveN * 32 + nt * 16 + lrow) * LDS_LD + lhi * 8];
      v8h b0 = *(const v8h*)p;
      v8h b1 = *(const v8h*)(p + 16);
      bfr[nt] = __builtin_shufflevector(b0, b1, 0, 1, 2, 3, 4, 5, 6, 7,
                                        8, 9, 10, 11, 12, 13, 14, 15);
    }
#pragma unroll
    for (int mt = 0; mt < 4; ++mt)
#pragma unroll
      for (int nt = 0; nt < 2; ++nt)
        acc[mt][nt] = __builtin_amdgcn_wmma_f32_16x16x32_f16(
            false, af[mt], false, bfr[nt], (short)0, acc[mt][nt], false, false);

#if !USE_ASYNC
    if (more) {
      const int nxt = 1 - cur;
      *(v8h*)&As[nxt][sO0] = ra0;  *(v8h*)&As[nxt][sO1] = ra1;
      *(v8h*)&Bs[nxt][sO0] = rb0;  *(v8h*)&Bs[nxt][sO1] = rb1;
    }
#endif
  }

  // epilogue: C/D layout -> VGPR r: M = r + 8*lhi, N = lrow
#pragma unroll
  for (int mt = 0; mt < 4; ++mt) {
#pragma unroll
    for (int nt = 0; nt < 2; ++nt) {
      const int gn  = bn0 + waveN * 32 + nt * 16 + lrow;
      const int gm0 = bm0 + waveM * 64 + mt * 16 + lhi * 8;
      float bv = 0.f;
      if (MODE == 0 || MODE == 2) bv = bias[gn];
#pragma unroll
      for (int r = 0; r < 8; ++r) {
        const size_t idx = (size_t)(gm0 + r) * ldo + gn;
        float v = acc[mt][nt][r];
        if (MODE == 0) {
          v += bv; v = v > 0.f ? v : 0.f;
          outH[idx] = (_Float16)v;
        } else if (MODE == 1) {
          outF[idx] += scale * v;
        } else {
          outF[idx] = v + bv;
        }
      }
    }
  }
}

// ---------------------------------------------------------------------------
extern "C" void kernel_launch(void* const* d_in, const int* in_sizes, int n_in,
                              void* d_out, int out_size, void* d_ws, size_t ws_size,
                              hipStream_t stream) {
  (void)in_sizes; (void)n_in; (void)out_size; (void)ws_size;
  const float* x  = (const float*)d_in[0];
  const float* W1 = (const float*)d_in[1];
  const float* b1 = (const float*)d_in[2];
  const float* W2 = (const float*)d_in[3];
  const float* b2 = (const float*)d_in[4];
  const float* Wf = (const float*)d_in[5];
  const float* bf = (const float*)d_in[6];
  float* out = (float*)d_out;

  char* ws = (char*)d_ws;
  const size_t MB = 1024ull * 1024ull;
  _Float16* xh   = (_Float16*)(ws + 0);        //  8 MB: x as f16
  _Float16* W1T  = (_Float16*)(ws + 8 * MB);   // 32 MB: W1^T f16 [E][H][D]
  _Float16* W2T  = (_Float16*)(ws + 40 * MB);  // 32 MB: W2^T f16 [E][O][H]
  _Float16* WfT  = (_Float16*)(ws + 72 * MB);  //  2 MB: Wf^T f16 [P][O]
  _Float16* Hh   = (_Float16*)(ws + 74 * MB);  // 16 MB: per-expert H f16
  float*    ybar = (float*)   (ws + 90 * MB);  // 16 MB: mean-y accumulator f32
  _Float16* ybh  = (_Float16*)(ws + 106 * MB); //  8 MB: ybar as f16

  // precision/layout passes
  k_f32_to_f16<<<(N_TOK * D_IN / 4 + 255) / 256, 256, 0, stream>>>(x, xh, N_TOK * D_IN / 4);
  dim3 tb(32, 8);
  k_transpose_convert<<<dim3(D_HID / 32, D_IN / 32, N_EXP), tb, 0, stream>>>(W1, W1T, D_IN, D_HID);
  k_transpose_convert<<<dim3(D_OUT / 32, D_HID / 32, N_EXP), tb, 0, stream>>>(W2, W2T, D_HID, D_OUT);
  k_transpose_convert<<<dim3(D_OUT / 32, D_OUT / 32, 1), tb, 0, stream>>>(Wf, WfT, D_OUT, D_OUT);
  k_bias_init<<<(N_TOK * D_OUT + 255) / 256, 256, 0, stream>>>(b2, ybar, N_TOK * D_OUT);

  // expert loop: H = relu(x W1e + b1e); ybar += (1/8) * H W2e
  for (int e = 0; e < N_EXP; ++e) {
    k_gemm<0><<<dim3(D_HID / 128, N_TOK / 128), 256, 0, stream>>>(
        xh, D_IN, W1T + (size_t)e * D_HID * D_IN, D_IN, b1 + e * D_HID,
        Hh, nullptr, D_HID, D_IN, 1.f);
    k_gemm<1><<<dim3(D_OUT / 128, N_TOK / 128), 256, 0, stream>>>(
        Hh, D_HID, W2T + (size_t)e * D_OUT * D_HID, D_HID, nullptr,
        nullptr, ybar, D_OUT, D_HID, 1.0f / N_EXP);
  }

  // out = ybar Wf + bf  (final linear commutes with the expert mean)
  k_f32_to_f16<<<(N_TOK * D_OUT / 4 + 255) / 256, 256, 0, stream>>>(ybar, ybh, N_TOK * D_OUT / 4);
  k_gemm<2><<<dim3(D_OUT / 128, N_TOK / 128), 256, 0, stream>>>(
      ybh, D_OUT, WfT, D_OUT, bf, nullptr, out, D_OUT, D_OUT, 1.f);
}
